// GNNEncoder_61040075211164
// MI455X (gfx1250) — matrix operand order repaired and checked by reading, not compiled, not run on hardware
//
#include <hip/hip_runtime.h>
#include <hip/hip_bf16.h>

typedef __attribute__((ext_vector_type(16))) _Float16 v16h;
typedef __attribute__((ext_vector_type(8)))  _Float16 v8h;
typedef __attribute__((ext_vector_type(8)))  float    v8f;

#define H 64
#define BN_EPS 1e-5f
#define EDGE_WAVES 4

static __device__ __forceinline__ v16h join16(v8h lo, v8h hi) {
  v16h r;
#pragma unroll
  for (int j = 0; j < 8; ++j) { r[j] = lo[j]; r[8 + j] = hi[j]; }
  return r;
}

// float atomic-max via signed-max (v>=0) / unsigned-min (v<0); agg init = -inf
static __device__ __forceinline__ void atomicMaxFloat(float* p, float v) {
  if (v >= 0.0f) atomicMax((int*)p, __float_as_int(v));
  else           atomicMin((unsigned int*)p, __float_as_uint(v));
}

// ---------------- weight prep: f32 [K][N] -> f16 transposed [N][K] ------------
__global__ void k_prep_w(const float* __restrict__ w1a, const float* __restrict__ w1b,
                         const float* __restrict__ w1c,
                         const float* __restrict__ w2a, const float* __restrict__ w2b,
                         const float* __restrict__ w2c,
                         _Float16* __restrict__ w1T, _Float16* __restrict__ w2T) {
  int gid = blockIdx.x * blockDim.x + threadIdx.x;
  const int W1 = 2 * H * H;   // 8192 per layer, [128][64]
  const int W2 = H * H;       // 4096 per layer, [64][64]
  if (gid < 3 * W1) {
    int l = gid / W1, r = gid % W1, k = r / H, n = r % H;
    const float* w = (l == 0) ? w1a : (l == 1) ? w1b : w1c;
    w1T[(size_t)l * W1 + n * (2 * H) + k] = (_Float16)w[k * H + n];
  } else {
    gid -= 3 * W1;
    if (gid < 3 * W2) {
      int l = gid / W2, r = gid % W2, k = r / H, n = r % H;
      const float* w = (l == 0) ? w2a : (l == 1) ? w2b : w2c;
      w2T[(size_t)l * W2 + n * H + k] = (_Float16)w[k * H + n];
    }
  }
}

// ---------------- input projection: x3[N,3] @ inW[3,64] + inB ----------------
__global__ void k_input_proj(const float* __restrict__ x3, const float* __restrict__ inW,
                             const float* __restrict__ inB,
                             float* __restrict__ x, _Float16* __restrict__ xh, int N) {
  int gid = blockIdx.x * blockDim.x + threadIdx.x;
  if (gid >= N * H) return;
  int n = gid >> 6, f = gid & 63;
  float v = inB[f] + x3[n * 3 + 0] * inW[f]
                   + x3[n * 3 + 1] * inW[H + f]
                   + x3[n * 3 + 2] * inW[2 * H + f];
  x[gid] = v;
  xh[gid] = (_Float16)v;
}

// ------- fill agg with -inf (scatter-max identity) and zero BN stats ---------
__global__ void k_fill(float* __restrict__ agg, float* __restrict__ stats, int n) {
  int gid = blockIdx.x * blockDim.x + threadIdx.x;
  if (gid < 2 * H) stats[gid] = 0.0f;
  int stride = gridDim.x * blockDim.x;
  for (int i = gid; i < n; i += stride) agg[i] = -__builtin_huge_valf();
}

// ---------------- edge MLP: WMMA f16, 16 edges per wave tile -----------------
__global__ __launch_bounds__(EDGE_WAVES * 32) void k_edge_mlp(
    const _Float16* __restrict__ xh,
    const int* __restrict__ srcIdx, const int* __restrict__ dstIdx,
    const _Float16* __restrict__ w1T,  // [64][128] f16 (N-major)
    const float* __restrict__ b1,
    const _Float16* __restrict__ w2T,  // [64][64]  f16 (N-major)
    const float* __restrict__ b2,
    float* __restrict__ agg, int numTiles, int E) {
  __shared__ __align__(64) _Float16 w1s[H * 2 * H];           // 16 KB
  __shared__ __align__(64) _Float16 w2s[H * H];               // 8 KB
  __shared__ __align__(64) _Float16 hls[EDGE_WAVES][16 * H];  // 8 KB

  const int tid = threadIdx.x;
  for (int i = tid; i < (H * 2 * H) / 8; i += blockDim.x)
    ((v8h*)w1s)[i] = ((const v8h*)w1T)[i];
  for (int i = tid; i < (H * H) / 8; i += blockDim.x)
    ((v8h*)w2s)[i] = ((const v8h*)w2T)[i];
  __syncthreads();

  const int wave = tid >> 5;
  const int lane = tid & 31;
  const int hlf  = lane >> 4;   // 0: lanes 0-15, 1: lanes 16-31
  const int l15  = lane & 15;
  const int o1   = hlf * 8;     // A-layout chunk offset within a 32-wide K window
  _Float16* hrow = &hls[wave][0];

  const int wgid  = blockIdx.x * EDGE_WAVES + wave;
  const int wstep = gridDim.x * EDGE_WAVES;

  for (int tile = wgid; tile < numTiles; tile += wstep) {
    const int ebase = tile * 16;
    int arow = ebase + l15; if (arow >= E) arow = E - 1;
    const _Float16* xi = xh + (long long)dstIdx[arow] * H;  // x_i = x[dst]
    const _Float16* xj = xh + (long long)srcIdx[arow] * H;  // x_j = x[src]

    // ef = [xi | xj - xi], A fragments (16x32 f16 each) loaded straight from global
    v16h A[4];
    A[0] = join16(*(const v8h*)(xi + o1),      *(const v8h*)(xi + o1 + 16));
    A[1] = join16(*(const v8h*)(xi + 32 + o1), *(const v8h*)(xi + 32 + o1 + 16));
    A[2] = join16(*(const v8h*)(xj + o1),      *(const v8h*)(xj + o1 + 16)) - A[0];
    A[3] = join16(*(const v8h*)(xj + 32 + o1), *(const v8h*)(xj + 32 + o1 + 16)) - A[1];

    // GEMM1: [16,128] @ [128,64] -> relu(+b1) -> h tile in LDS
#pragma unroll
    for (int n = 0; n < 4; ++n) {
      v8f acc = {0.f, 0.f, 0.f, 0.f, 0.f, 0.f, 0.f, 0.f};
#pragma unroll
      for (int f = 0; f < 4; ++f) {
        v16h B = join16(*(const v8h*)&w1s[(n * 16 + l15) * (2 * H) + f * 32 + o1],
                        *(const v8h*)&w1s[(n * 16 + l15) * (2 * H) + f * 32 + o1 + 16]);
        acc = __builtin_amdgcn_wmma_f32_16x16x32_f16(false, A[f], false, B,
                                                     (short)0, acc, false, false);
      }
      const int col = n * 16 + l15;
      const float bb = b1[col];
#pragma unroll
      for (int r = 0; r < 8; ++r) {
        float v = acc[r] + bb;
        v = v > 0.f ? v : 0.f;
        hrow[(hlf * 8 + r) * H + col] = (_Float16)v;   // D layout -> row-major LDS
      }
    }

    // re-stripe h into A layout for GEMM2
    v16h A2[2];
#pragma unroll
    for (int g = 0; g < 2; ++g)
      A2[g] = join16(*(const v8h*)(hrow + l15 * H + g * 32 + o1),
                     *(const v8h*)(hrow + l15 * H + g * 32 + o1 + 16));

    long long nodes[8];
#pragma unroll
    for (int r = 0; r < 8; ++r) {
      int e = ebase + hlf * 8 + r;
      nodes[r] = (e < E) ? (long long)dstIdx[e] : -1;
    }

    // GEMM2: [16,64] @ [64,64] -> +b2 -> scatter-max into agg[dst]
#pragma unroll
    for (int n2 = 0; n2 < 4; ++n2) {
      v8f acc = {0.f, 0.f, 0.f, 0.f, 0.f, 0.f, 0.f, 0.f};
#pragma unroll
      for (int g = 0; g < 2; ++g) {
        v16h B = join16(*(const v8h*)&w2s[(n2 * 16 + l15) * H + g * 32 + o1],
                        *(const v8h*)&w2s[(n2 * 16 + l15) * H + g * 32 + o1 + 16]);
        acc = __builtin_amdgcn_wmma_f32_16x16x32_f16(false, A2[g], false, B,
                                                     (short)0, acc, false, false);
      }
      const int col = n2 * 16 + l15;
      const float bb = b2[col];
#pragma unroll
      for (int r = 0; r < 8; ++r) {
        if (nodes[r] >= 0) atomicMaxFloat(agg + nodes[r] * H + col, acc[r] + bb);
      }
    }
  }
}

// ------- agg cleanup + residual + per-feature sum/sumsq (BN stats) -----------
// mode 0: x = clean(agg); mode 1: x = x + clean(agg); mode 2: read-only pooling
__global__ __launch_bounds__(256) void k_finish(float* __restrict__ x,
                                                const float* __restrict__ agg,
                                                float* __restrict__ stats,
                                                int total, int mode) {
  __shared__ float ssum[H], ssq[H];
  int tid = threadIdx.x;
  if (tid < H) { ssum[tid] = 0.f; ssq[tid] = 0.f; }
  __syncthreads();
  int gid = blockIdx.x * blockDim.x + tid;
  int stride = gridDim.x * blockDim.x;  // launched as multiple of 64
  int f = gid & 63;
  float ls = 0.f, lq = 0.f;
  for (int i = gid; i < total; i += stride) {
    float v;
    if (mode == 2) {
      v = x[i];
    } else {
      float a = agg[i];
      a = __builtin_isfinite(a) ? a : 0.f;  // empty nodes: -inf -> 0
      v = mode ? (x[i] + a) : a;
      x[i] = v;
    }
    ls += v; lq += v * v;
  }
  atomicAdd(&ssum[f], ls);
  atomicAdd(&ssq[f], lq);
  __syncthreads();
  if (tid < H) { atomicAdd(&stats[tid], ssum[tid]); atomicAdd(&stats[H + tid], ssq[tid]); }
}

// ---------------- batch-norm + relu, refresh f16 mirror ----------------------
__global__ void k_bn(float* __restrict__ x, _Float16* __restrict__ xh,
                     const float* __restrict__ stats, const float* __restrict__ gma,
                     const float* __restrict__ bta, int total, float invN) {
  int gid = blockIdx.x * blockDim.x + threadIdx.x;
  if (gid >= total) return;
  int f = gid & 63;
  float mu  = stats[f] * invN;
  float var = stats[H + f] * invN - mu * mu;
  float v = (x[gid] - mu) * rsqrtf(var + BN_EPS) * gma[f] + bta[f];
  v = v > 0.f ? v : 0.f;
  x[gid] = v;
  xh[gid] = (_Float16)v;
}

// ---------------- head: mean-pool -> 64->64 relu -> 64->32 -------------------
__global__ __launch_bounds__(64) void k_head(const float* __restrict__ stats,
                                             const float* __restrict__ w1,
                                             const float* __restrict__ b1,
                                             const float* __restrict__ w2,
                                             const float* __restrict__ b2,
                                             float* __restrict__ out, float invN) {
  __shared__ float z[H], h[H];
  int f = threadIdx.x;
  z[f] = stats[f] * invN;
  __syncthreads();
  float v = b1[f];
  for (int k = 0; k < H; ++k) v += z[k] * w1[k * H + f];
  h[f] = v > 0.f ? v : 0.f;
  __syncthreads();
  if (f < 32) {
    float o = b2[f];
    for (int k = 0; k < H; ++k) o += h[k] * w2[k * 32 + f];
    out[f] = o;
  }
}

extern "C" void kernel_launch(void* const* d_in, const int* in_sizes, int n_in,
                              void* d_out, int out_size, void* d_ws, size_t ws_size,
                              hipStream_t stream) {
  (void)n_in; (void)out_size; (void)ws_size;
  const int N = in_sizes[0] / 3;
  const int E = in_sizes[1] / 2;
  const float* x3     = (const float*)d_in[0];
  const int*   srcIdx = (const int*)d_in[1];       // edge_index[0] = src
  const int*   dstIdx = srcIdx + E;                // edge_index[1] = dst
  const float* inW    = (const float*)d_in[2];
  const float* inB    = (const float*)d_in[3];
  const float* o_w1   = (const float*)d_in[22];
  const float* o_b1   = (const float*)d_in[23];
  const float* o_w2   = (const float*)d_in[24];
  const float* o_b2   = (const float*)d_in[25];

  // workspace carve (256B aligned): ~64 MB total — L2 resident on MI455X
  char* wsp = (char*)d_ws;
  auto carve = [&](size_t bytes) {
    char* p = wsp; wsp += (bytes + 255) & ~(size_t)255; return p;
  };
  const size_t NH = (size_t)N * H;
  float*    x     = (float*)carve(NH * 4);
  _Float16* xh    = (_Float16*)carve(NH * 2);
  float*    agg   = (float*)carve(NH * 4);
  float*    stats = (float*)carve(2 * H * 4);
  _Float16* w1T   = (_Float16*)carve((size_t)3 * 2 * H * H * 2);
  _Float16* w2T   = (_Float16*)carve((size_t)3 * H * H * 2);

  const int total = (int)NH;
  const float invN = 1.0f / (float)N;

  {
    int tot = 3 * (2 * H * H) + 3 * (H * H);
    k_prep_w<<<(tot + 255) / 256, 256, 0, stream>>>(
        (const float*)d_in[4], (const float*)d_in[10], (const float*)d_in[16],
        (const float*)d_in[6], (const float*)d_in[12], (const float*)d_in[18],
        w1T, w2T);
  }
  k_input_proj<<<(total + 255) / 256, 256, 0, stream>>>(x3, inW, inB, x, xh, N);

  const int numTiles = (E + 15) / 16;
  int edgeBlocks = (numTiles + EDGE_WAVES - 1) / EDGE_WAVES;
  if (edgeBlocks > 4096) edgeBlocks = 4096;

  for (int l = 0; l < 3; ++l) {
    const int base = 4 + 6 * l;  // w1, b1, w2, b2, bn_g, bn_b
    k_fill<<<1024, 256, 0, stream>>>(agg, stats, total);
    k_edge_mlp<<<edgeBlocks, EDGE_WAVES * 32, 0, stream>>>(
        xh, srcIdx, dstIdx,
        w1T + (size_t)l * 2 * H * H, (const float*)d_in[base + 1],
        w2T + (size_t)l * H * H,     (const float*)d_in[base + 3],
        agg, numTiles, E);
    k_finish<<<1024, 256, 0, stream>>>(x, agg, stats, total, (l > 0) ? 1 : 0);
    k_bn<<<(total + 255) / 256, 256, 0, stream>>>(
        x, xh, stats, (const float*)d_in[base + 4], (const float*)d_in[base + 5],
        total, invN);
  }

  // global mean pool + output MLP
  k_fill<<<1, 128, 0, stream>>>(agg, stats, 0);  // zero stats only
  k_finish<<<1024, 256, 0, stream>>>(x, agg, stats, total, 2);
  k_head<<<1, 64, 0, stream>>>(stats, o_w1, o_b1, o_w2, o_b2, (float*)d_out, invN);
}
